// AttentiveDecoderNetwork_42434276884537
// MI455X (gfx1250) — matrix operand order repaired and checked by reading, not compiled, not run on hardware
//
#include <hip/hip_runtime.h>
#include <hip/hip_bf16.h>
#include <math.h>

// ---------------------------------------------------------------------------
// Problem constants (from the reference)
// ---------------------------------------------------------------------------
#define BB   64
#define SS   128
#define TT   64
#define HH   1024
#define EE   512
#define VV   16000
#define ENCC 2048          // 2*H
#define XD   2560          // E + ENC
#define G3   3072          // 3*H
#define SOS_TOK 1

typedef __attribute__((ext_vector_type(16))) __bf16 v16bf;
typedef __attribute__((ext_vector_type(8)))  __bf16 v8bf;
typedef __attribute__((ext_vector_type(4)))  __bf16 v4bf;
typedef __attribute__((ext_vector_type(8)))  float  v8f;
typedef __attribute__((ext_vector_type(4)))  float  v4f;

// ---------------------------------------------------------------------------
// Cast an f32 sub-matrix (rows x cols, leading dim srcld, column offset col0)
// into a contiguous bf16 matrix. cols is a multiple of 4 -> vectorized.
// ---------------------------------------------------------------------------
__global__ __launch_bounds__(256)
void cast_mat_bf16(const float* __restrict__ src, int srcld, int col0,
                   __bf16* __restrict__ dst, int cols, int total4)
{
    int i = blockIdx.x * 256 + threadIdx.x;   // index in float4 units
    if (i >= total4) return;
    int cols4 = cols >> 2;
    int r  = i / cols4;
    int c4 = i - r * cols4;
    v4f f = *(const v4f*)(src + (long long)r * srcld + col0 + c4 * 4);
    v4bf o;
#pragma unroll
    for (int e = 0; e < 4; ++e) o[e] = (__bf16)f[e];
    *(v4bf*)(dst + (long long)r * cols + c4 * 4) = o;
}

// ---------------------------------------------------------------------------
// WMMA bf16 GEMM:  out[m,n] = sum_k A[m,k] * W[n,k] (+ bias[n])
//   A   : f32, M x K row-major, leading dim lda
//   W   : bf16, N x K row-major, leading dim ldw (pre-cast weights)
//   out : f32, row stride ldo
// Block = 256 threads (8 waves), computes a 64(M) x 128(N) tile:
//   wave -> 2 m-tiles at (wave&1)*32 (+0,+16)
//        x  2 n-tiles at (wave>>1)*32 (+0,+16)
//   grid.x = N/128, grid.y = M/64.  K must be a multiple of 32.
// A tile (64x32 f32->bf16) staged into double-buffered LDS: one barrier
// per k-step; 4 wmma per wave per k-step.
// ---------------------------------------------------------------------------
__global__ __launch_bounds__(256)
void gemm_bf16_wmma(const float* __restrict__ A, int lda,
                    const __bf16* __restrict__ W, int ldw,
                    const float* __restrict__ bias,
                    float* __restrict__ out, long long ldo,
                    int K)
{
    __shared__ __bf16 As[2][64 * 32];    // 8 KB double-buffered A tile

    const int tid   = threadIdx.x;
    const int wave  = tid >> 5;
    const int lane  = tid & 31;
    const int mhalf = wave & 1;          // 2 m-tiles at mhalf*32, mhalf*32+16
    const int npair = wave >> 1;         // 2 n-tiles at npair*32, npair*32+16
    const int m0    = blockIdx.y * 64;
    const int n0    = blockIdx.x * 128 + npair * 32;

    const int row = lane & 15;           // A row within tile / B column
    const int kh  = lane >> 4;           // K-half selector

    // --- fixed staging slots: thread handles rows sr and sr+32, 4 cols ---
    const int sr  = tid >> 3;            // 0..31
    const int sc0 = (tid & 7) * 4;       // 0,4,...,28
    const float* pA0 = A + (long long)(m0 + sr) * lda + sc0;
    const float* pA1 = pA0 + (long long)32 * lda;
    __bf16* sd0 = &As[0][sr * 32 + sc0];
    __bf16* sd1 = &As[0][(sr + 32) * 32 + sc0];

    // --- fixed fragment base pointers ---
    const __bf16* wb0 = W + (long long)(n0 + row) * ldw + 16 * kh;   // n-tile 0
    const __bf16* wb1 = wb0 + (long long)16 * ldw;                   // n-tile 1
    const __bf16* af0 = &As[0][(mhalf * 32 + row) * 32 + 16 * kh];   // m-tile 0
    // m-tile 1 is af0 + 16*32

    v8f acc00 = {0.f,0.f,0.f,0.f,0.f,0.f,0.f,0.f};
    v8f acc01 = acc00, acc10 = acc00, acc11 = acc00;

    // stage one 64x32 tile (f32 -> packed bf16) into buffer b
    auto stage = [&](int kk, int b) {
        v4f f0 = *(const v4f*)(pA0 + kk);
        v4f f1 = *(const v4f*)(pA1 + kk);
        v4bf o0, o1;
#pragma unroll
        for (int e = 0; e < 4; ++e) { o0[e] = (__bf16)f0[e]; o1[e] = (__bf16)f1[e]; }
        *(v4bf*)(sd0 + b * (64 * 32)) = o0;
        *(v4bf*)(sd1 + b * (64 * 32)) = o1;
    };

    stage(0, 0);                          // prologue: first tile into buf 0
    int buf = 0;

    for (int k0 = 0; k0 < K; k0 += 32) {
        __syncthreads();                  // staged tile visible to all waves
        if (k0 + 32 < K) stage(k0 + 32, buf ^ 1);   // prefetch next tile

        // A fragments (16x32 bf16 each): contiguous 16B LDS reads per lane
        const __bf16* ab = af0 + buf * (64 * 32);
        v16bf afrag0, afrag1;
        {
            v8bf l0 = *(const v8bf*)(ab);
            v8bf h0v = *(const v8bf*)(ab + 8);
            v8bf l1 = *(const v8bf*)(ab + 16 * 32);
            v8bf h1v = *(const v8bf*)(ab + 16 * 32 + 8);
#pragma unroll
            for (int e = 0; e < 8; ++e) {
                afrag0[e] = l0[e]; afrag0[8 + e] = h0v[e];
                afrag1[e] = l1[e]; afrag1[8 + e] = h1v[e];
            }
        }

        // B fragments: 16 contiguous k per lane == fragment order directly
        v16bf bfrag0 = *(const v16bf*)(wb0 + k0);
        v16bf bfrag1 = *(const v16bf*)(wb1 + k0);

        acc00 = __builtin_amdgcn_wmma_f32_16x16x32_bf16(
                    false, afrag0, false, bfrag0, (short)0, acc00, false, false);
        acc01 = __builtin_amdgcn_wmma_f32_16x16x32_bf16(
                    false, afrag0, false, bfrag1, (short)0, acc01, false, false);
        acc10 = __builtin_amdgcn_wmma_f32_16x16x32_bf16(
                    false, afrag1, false, bfrag0, (short)0, acc10, false, false);
        acc11 = __builtin_amdgcn_wmma_f32_16x16x32_bf16(
                    false, afrag1, false, bfrag1, (short)0, acc11, false, false);

        buf ^= 1;
    }

    // C/D layout: VGPR r, lanes 0-15 -> M=r, lanes 16-31 -> M=8+r
    const int nA = n0 + row;
    const int nB = n0 + 16 + row;
    const float bvA = bias ? bias[nA] : 0.f;
    const float bvB = bias ? bias[nB] : 0.f;
    const int mbase0 = m0 + mhalf * 32 + 8 * kh;
    const int mbase1 = mbase0 + 16;
#pragma unroll
    for (int r = 0; r < 8; ++r) {
        out[(long long)(mbase0 + r) * ldo + nA] = acc00[r] + bvA;
        out[(long long)(mbase0 + r) * ldo + nB] = acc01[r] + bvB;
        out[(long long)(mbase1 + r) * ldo + nA] = acc10[r] + bvA;
        out[(long long)(mbase1 + r) * ldo + nB] = acc11[r] + bvB;
    }
}

// ---------------------------------------------------------------------------
// Attention: per batch row b (one workgroup):
//   scores[s] = sum_h v[h] * tanh(q[b,h] + enc_proj[b,s,h])
//   alpha    = softmax(scores)
//   ctx[b,d] = sum_s alpha[s] * enc[b,s,d]
// ---------------------------------------------------------------------------
__global__ __launch_bounds__(256)
void attn_kernel(const float* __restrict__ q,
                 const float* __restrict__ enc_proj,
                 const float* __restrict__ vvec,
                 const float* __restrict__ enc,
                 float* __restrict__ ctx)
{
    const int b   = blockIdx.x;
    const int tid = threadIdx.x;
    const int wave = tid >> 5, lane = tid & 31;

    __shared__ float qs[HH];
    __shared__ float sc[SS];
    __shared__ float smax, ssum;

    for (int h4 = tid; h4 < HH / 4; h4 += 256)
        *(v4f*)&qs[h4 * 4] = *(const v4f*)(q + b * HH + h4 * 4);
    __syncthreads();

    // energy + dot with v: one wave per source position, float4-strided
    for (int s = wave; s < SS; s += 8) {
        const float* ep = enc_proj + ((long long)(b * SS + s)) * HH;
        float p = 0.f;
        for (int h4 = lane; h4 < HH / 4; h4 += 32) {
            v4f e  = *(const v4f*)(ep + h4 * 4);
            v4f qv = *(const v4f*)&qs[h4 * 4];
            v4f vv = *(const v4f*)(vvec + h4 * 4);
#pragma unroll
            for (int e2 = 0; e2 < 4; ++e2)
                p += vv[e2] * tanhf(qv[e2] + e[e2]);
        }
#pragma unroll
        for (int off = 16; off > 0; off >>= 1)
            p += __shfl_xor(p, off, 32);
        if (lane == 0) sc[s] = p;
    }
    __syncthreads();

    if (tid == 0) {
        float mx = sc[0];
        for (int s = 1; s < SS; ++s) mx = fmaxf(mx, sc[s]);
        smax = mx;
    }
    __syncthreads();
    if (tid < SS) sc[tid] = expf(sc[tid] - smax);
    __syncthreads();
    if (tid == 0) {
        float sum = 0.f;
        for (int s = 0; s < SS; ++s) sum += sc[s];
        ssum = sum;
    }
    __syncthreads();
    if (tid < SS) sc[tid] /= ssum;
    __syncthreads();

    // ctx = alpha @ encoder_outputs  (float4 columns per thread)
    for (int d4 = tid; d4 < ENCC / 4; d4 += 256) {
        const float* e = enc + (long long)b * SS * ENCC + d4 * 4;
        v4f acc = {0.f, 0.f, 0.f, 0.f};
        for (int s = 0; s < SS; ++s) {
            v4f ev = *(const v4f*)(e + (long long)s * ENCC);
            float a = sc[s];
#pragma unroll
            for (int e2 = 0; e2 < 4; ++e2) acc[e2] += a * ev[e2];
        }
        *(v4f*)&ctx[b * ENCC + d4 * 4] = acc;
    }
}

// ---------------------------------------------------------------------------
// x = concat(emb[tok], ctx)   (64 x 2560 f32)
// ---------------------------------------------------------------------------
__global__ __launch_bounds__(256)
void build_x(const int* __restrict__ target, int t,
             const float* __restrict__ emb,
             const float* __restrict__ ctx,
             float* __restrict__ x)
{
    int i4 = blockIdx.x * 256 + threadIdx.x;       // float4 units
    if (i4 >= BB * XD / 4) return;
    int b = i4 / (XD / 4), k4 = i4 - b * (XD / 4);
    int k = k4 * 4;
    int tok = (t == 0) ? SOS_TOK : target[b * TT + (t - 1)];
    v4f val = (k < EE) ? *(const v4f*)(emb + (long long)tok * EE + k)
                       : *(const v4f*)(ctx + b * ENCC + (k - EE));
    *(v4f*)(x + (long long)b * XD + k) = val;
}

// ---------------------------------------------------------------------------
// GRU gate fusion: h_new = (1-z)*n + z*h
// ---------------------------------------------------------------------------
__global__ __launch_bounds__(256)
void gru_update(const float* __restrict__ gi, const float* __restrict__ gh,
                const float* __restrict__ h, float* __restrict__ hnew)
{
    int i = blockIdx.x * 256 + threadIdx.x;
    if (i >= BB * HH) return;
    int b = i >> 10, j = i & (HH - 1);
    const float* gib = gi + b * G3;
    const float* ghb = gh + b * G3;
    float r = 1.f / (1.f + expf(-(gib[j]          + ghb[j])));
    float z = 1.f / (1.f + expf(-(gib[HH + j]     + ghb[HH + j])));
    float n = tanhf(gib[2 * HH + j] + r * ghb[2 * HH + j]);
    hnew[i] = (1.f - z) * n + z * h[i];
}

// ---------------------------------------------------------------------------
// Host orchestration
// ---------------------------------------------------------------------------
extern "C" void kernel_launch(void* const* d_in, const int* in_sizes, int n_in,
                              void* d_out, int out_size, void* d_ws, size_t ws_size,
                              hipStream_t stream)
{
    const float* enc      = (const float*)d_in[0];   // (B,S,ENC)
    const float* enc_h    = (const float*)d_in[1];   // (1,B,H)
    const int*   target   = (const int*)  d_in[2];   // (B,T)
    const float* emb      = (const float*)d_in[3];   // (V,E)
    const float* Wa       = (const float*)d_in[4];   // (H, H+ENC)
    const float* ba       = (const float*)d_in[5];   // (H)
    const float* vvec     = (const float*)d_in[6];   // (H)
    const float* W_ih     = (const float*)d_in[7];   // (3H, E+ENC)
    const float* b_ih     = (const float*)d_in[8];
    const float* W_hh     = (const float*)d_in[9];   // (3H, H)
    const float* b_hh     = (const float*)d_in[10];
    const float* Wo       = (const float*)d_in[11];  // (V, H)
    const float* bo       = (const float*)d_in[12];
    float*       out      = (float*)d_out;           // (B,T,V)

    // ---- workspace carve-up (256B aligned) ----
    char* ws = (char*)d_ws;
    size_t off = 0;
    auto alloc = [&](size_t bytes) -> char* {
        char* p = ws + off;
        off = (off + bytes + 255) & ~(size_t)255;
        return p;
    };
    __bf16* Wq_bf  = (__bf16*)alloc((size_t)HH * HH   * 2);
    __bf16* Wae_bf = (__bf16*)alloc((size_t)HH * ENCC * 2);
    __bf16* Wih_bf = (__bf16*)alloc((size_t)G3 * XD   * 2);
    __bf16* Whh_bf = (__bf16*)alloc((size_t)G3 * HH   * 2);
    __bf16* Wo_bf  = (__bf16*)alloc((size_t)VV * HH   * 2);
    float*  encp   = (float*)alloc((size_t)BB * SS * HH * 4);  // enc_proj
    float*  h0     = (float*)alloc((size_t)BB * HH * 4);
    float*  h1     = (float*)alloc((size_t)BB * HH * 4);
    float*  qbuf   = (float*)alloc((size_t)BB * HH * 4);
    float*  ctx    = (float*)alloc((size_t)BB * ENCC * 4);
    float*  xbuf   = (float*)alloc((size_t)BB * XD * 4);
    float*  gi     = (float*)alloc((size_t)BB * G3 * 4);
    float*  gh     = (float*)alloc((size_t)BB * G3 * 4);
    (void)ws_size; (void)n_in; (void)in_sizes; (void)out_size;

    const int TPB = 256;
    auto blocks = [](int total) { return (total + 255) / 256; };

    // ---- one-time weight casts to bf16 (counts in float4 units) ----
    cast_mat_bf16<<<blocks(HH * HH / 4),   TPB, 0, stream>>>(Wa,  HH + ENCC, 0,  Wq_bf,  HH,   HH * HH / 4);
    cast_mat_bf16<<<blocks(HH * ENCC / 4), TPB, 0, stream>>>(Wa,  HH + ENCC, HH, Wae_bf, ENCC, HH * ENCC / 4);
    cast_mat_bf16<<<blocks(G3 * XD / 4),   TPB, 0, stream>>>(W_ih, XD, 0,        Wih_bf, XD,   G3 * XD / 4);
    cast_mat_bf16<<<blocks(G3 * HH / 4),   TPB, 0, stream>>>(W_hh, HH, 0,        Whh_bf, HH,   G3 * HH / 4);
    cast_mat_bf16<<<blocks(VV * HH / 4),   TPB, 0, stream>>>(Wo,   HH, 0,        Wo_bf,  HH,   VV * HH / 4);

    // ---- enc_proj = einsum('bse,he->bsh'): (B*S) x H x ENC GEMM ----
    {
        dim3 g(HH / 128, (BB * SS) / 64);
        gemm_bf16_wmma<<<g, TPB, 0, stream>>>(enc, ENCC, Wae_bf, ENCC,
                                              nullptr, encp, (long long)HH, ENCC);
    }

    // ---- h <- encoder_hidden[0] ----
    hipMemcpyAsync(h0, enc_h, (size_t)BB * HH * 4, hipMemcpyDeviceToDevice, stream);

    float* hcur = h0;
    float* hnew = h1;

    for (int t = 0; t < TT; ++t) {
        // q = h @ Wa_q^T + ba           (64 x 1024 x 1024)
        {
            dim3 g(HH / 128, 1);
            gemm_bf16_wmma<<<g, TPB, 0, stream>>>(hcur, HH, Wq_bf, HH,
                                                  ba, qbuf, (long long)HH, HH);
        }
        // attention -> ctx
        attn_kernel<<<BB, TPB, 0, stream>>>(qbuf, encp, vvec, enc, ctx);

        // x = [emb[tok], ctx]
        build_x<<<blocks(BB * XD / 4), TPB, 0, stream>>>(target, t, emb, ctx, xbuf);

        // gi = x @ W_ih^T + b_ih        (64 x 3072 x 2560)
        {
            dim3 g(G3 / 128, 1);
            gemm_bf16_wmma<<<g, TPB, 0, stream>>>(xbuf, XD, Wih_bf, XD,
                                                  b_ih, gi, (long long)G3, XD);
        }
        // gh = h @ W_hh^T + b_hh        (64 x 3072 x 1024)
        {
            dim3 g(G3 / 128, 1);
            gemm_bf16_wmma<<<g, TPB, 0, stream>>>(hcur, HH, Whh_bf, HH,
                                                  b_hh, gh, (long long)G3, HH);
        }
        // GRU gates -> h_new
        gru_update<<<blocks(BB * HH), TPB, 0, stream>>>(gi, gh, hcur, hnew);

        // out[:, t, :] = h_new @ Wo^T + bo   (64 x 16000 x 1024)
        {
            dim3 g(VV / 128, 1);
            gemm_bf16_wmma<<<g, TPB, 0, stream>>>(hnew, HH, Wo_bf, HH,
                                                  bo, out + (long long)t * VV,
                                                  (long long)TT * VV, HH);
        }

        float* tmp = hcur; hcur = hnew; hnew = tmp;
    }
}
